// MyTransformer_47553877902054
// MI455X (gfx1250) — matrix operand order repaired and checked by reading, not compiled, not run on hardware
//
#include <hip/hip_runtime.h>

// ---------------------------------------------------------------------------
// Transformer forward (enc-dec, L=3) for MI455X / gfx1250, wave32 + WMMA bf16.
// GEMMs: v_wmma_f32_16x16x32_bf16, 128x128 block tiles, 8 waves, double-
// buffered LDS with register prefetch. LDS tiles are stored in *fragment
// order* so every v16bf operand is a single aligned 32B LDS read
// (2x ds_load_b128); global staging is float4 (global_load_b128); f32->bf16
// uses hardware packed converts (__builtin_convertvector -> v_cvt_pk_bf16).
// Flags/causal are template params so addressing folds at compile time.
// Sizes fixed: B=2 S=1024 D=1024 H=16 DH=64 DFF=4096 V=32000. WS: 88MB.
// ---------------------------------------------------------------------------

typedef __attribute__((ext_vector_type(16))) __bf16 v16bf;
typedef __attribute__((ext_vector_type(4)))  __bf16 bf4;
typedef __attribute__((ext_vector_type(4)))  float  vf4;
typedef __attribute__((ext_vector_type(8)))  float  v8f;

#define TS   1024
#define TB   2
#define TD   1024
#define TH   16
#define TDH  64
#define TDFF 4096
#define TV   32000
#define TL   3
#define TM   (TB*TS)

#define F_RELU 1
#define F_QKV  2   // B stored as [H, K, 64] (per-head stacked weights)
#define F_SWZ  4   // output row m=b*S+s written to row s*B+b (final logits)

__device__ __forceinline__ __bf16 f2bf(float f) { return (__bf16)f; }

__device__ __forceinline__ bf4 cvt_bf4(float4 v) {
  vf4 f = { v.x, v.y, v.z, v.w };
  return __builtin_convertvector(f, bf4);   // v_cvt_pk_bf16_f32 x2
}
// pack a float4 into 4 bf16 and store as one 8-byte LDS write
__device__ __forceinline__ void st_bf4(__bf16* dst, float4 v) {
  *(bf4*)dst = cvt_bf4(v);
}
// one whole fragment = 32 contiguous bytes at (base + lane*16) -> 2x b128
__device__ __forceinline__ v16bf ld_frag(const __bf16* base) {
  return *(const v16bf*)(base + (threadIdx.x & 31) * 16);
}

// Fragment-order permutation (this code's contiguous k order, not the ISA table order):
//  A (16x32, MxK): lane = (m&15) + 16*(k>>4),     slot = k&15
//  B (32x16, KxN): lane = (n&15) + 16*(k>>4),     slot = k&15
// Each 16x32 (A) / 32x16 (B) subtile occupies 512 bf16 = [32 lanes][16 slots].

// ---------------------------------------------------------------------------
// Embedding + positional encoding -> ENC and DEC [B*S, D]
// ---------------------------------------------------------------------------
__global__ __launch_bounds__(256) void embed_k(const int* __restrict__ inp,
                                               const float* __restrict__ PE,
                                               const float* __restrict__ emb,
                                               float* __restrict__ ENC,
                                               float* __restrict__ DEC) {
  const int row = blockIdx.x;            // b*S + s
  const int b = row >> 10, s = row & (TS - 1);
  const int tok = inp[s * TB + b];
  const int tid = threadIdx.x;
  const int d = tid * 4;
  float4 e = *(const float4*)(emb + (size_t)tok * TD + d);
  float4 p = *(const float4*)(PE + (size_t)s * TD + d);
  float4 v; v.x = e.x + p.x; v.y = e.y + p.y; v.z = e.z + p.z; v.w = e.w + p.w;
  *(float4*)(ENC + (size_t)row * TD + d) = v;
  *(float4*)(DEC + (size_t)row * TD + d) = v;
}

// ---------------------------------------------------------------------------
// GEMM: C[M,N] = A[M,K](f32) x B[K,N](f32) + bias, FLAGS: RELU/QKV/SWZ.
// Block 256 thr = 8 waves, tile 128x128, k-step 32; wave tile 64x32 (8 WMMA).
// Double-buffered fragment-order LDS + register prefetch of next tile.
// ---------------------------------------------------------------------------
template <int FLAGS>
__global__ __launch_bounds__(256) void gemm_k(const float* __restrict__ A,
                                              const float* __restrict__ Bw,
                                              const float* __restrict__ bias,
                                              float* __restrict__ C,
                                              int M, int N, int K) {
  __shared__ alignas(32) __bf16 sA[2][8 * 512];   // 8 subtiles of 16x32
  __shared__ alignas(32) __bf16 sB[2][8 * 512];   // 8 subtiles of 32x16

  const int tid = threadIdx.x;
  const int m0 = blockIdx.x * 128, n0 = blockIdx.y * 128;
  const int w = tid >> 5, wm = w >> 2, wn = w & 3;   // wave grid 2x4

  float4 ra[4], rb[4];

  auto load_tiles = [&](int k0) {
#pragma unroll
    for (int i = 0; i < 4; ++i) {
      const int ia = i * 256 + tid;
      const int m_l = ia >> 3, k4 = (ia & 7) * 4;
      ra[i] = *(const float4*)(A + (size_t)(m0 + m_l) * K + k0 + k4);
      const int k_l = ia >> 5, n4 = (ia & 31) * 4;
      const int n = n0 + n4, kk = k0 + k_l;
      size_t gi;
      if (FLAGS & F_QKV) gi = ((size_t)(n >> 6) * K + kk) * 64 + (n & 63);
      else               gi = (size_t)kk * N + n;
      rb[i] = *(const float4*)(Bw + gi);
    }
  };
  auto store_tiles = [&](int buf) {
#pragma unroll
    for (int i = 0; i < 4; ++i) {
      const int ia = i * 256 + tid;
      // A: 4-aligned k quad -> one lane, 4 contiguous slots
      const int m_l = ia >> 3, k4 = (ia & 7) * 4;
      const int asub = m_l >> 4;
      const int alane = (m_l & 15) + 16 * (k4 >> 4);
      const int aslot = k4 & 15;
      st_bf4(&sA[buf][asub * 512 + alane * 16 + aslot], ra[i]);
      // B: 4 consecutive n -> 4 lanes, same slot
      const int k_l = ia >> 5, n4 = (ia & 31) * 4;
      const int bsub = n4 >> 4;
      const int blane = (n4 & 15) + 16 * (k_l >> 4);
      const int bslot = k_l & 15;
      __bf16* p = &sB[buf][bsub * 512 + blane * 16 + bslot];
      bf4 bb = cvt_bf4(rb[i]);
      p[0]  = bb[0];
      p[16] = bb[1];
      p[32] = bb[2];
      p[48] = bb[3];
    }
  };

  v8f acc[4][2] = {};
  const int nT = K / 32;

  load_tiles(0);
  store_tiles(0);
  __syncthreads();

  for (int t = 0; t < nT; ++t) {
    const int cur = t & 1;
    if (t + 1 < nT) load_tiles((t + 1) * 32);   // prefetch into registers

    v16bf af[4], bf0, bf1;
#pragma unroll
    for (int mi = 0; mi < 4; ++mi)
      af[mi] = ld_frag(&sA[cur][(wm * 4 + mi) * 512]);
    bf0 = ld_frag(&sB[cur][(wn * 2 + 0) * 512]);
    bf1 = ld_frag(&sB[cur][(wn * 2 + 1) * 512]);
#pragma unroll
    for (int mi = 0; mi < 4; ++mi) {
      acc[mi][0] = __builtin_amdgcn_wmma_f32_16x16x32_bf16(
          false, af[mi], false, bf0, (short)0, acc[mi][0], false, false);
      acc[mi][1] = __builtin_amdgcn_wmma_f32_16x16x32_bf16(
          false, af[mi], false, bf1, (short)0, acc[mi][1], false, false);
    }
    if (t + 1 < nT) {
      store_tiles(1 - cur);
      __syncthreads();
    }
  }

  // epilogue: C/D layout -> lane n = lane&15, VGPR r -> m = r + 8*half
  const int lane = tid & 31, half = lane >> 4, l16 = lane & 15;
#pragma unroll
  for (int mi = 0; mi < 4; ++mi)
#pragma unroll
    for (int nj = 0; nj < 2; ++nj)
#pragma unroll
      for (int r = 0; r < 8; ++r) {
        const int m = m0 + wm * 64 + mi * 16 + r + 8 * half;
        const int n = n0 + wn * 32 + nj * 16 + l16;
        float v = acc[mi][nj][r] + bias[n];
        if (FLAGS & F_RELU) v = fmaxf(v, 0.0f);
        size_t oi;
        if (FLAGS & F_SWZ) {
          const int bb = m >> 10, ss = m & (TS - 1);
          oi = ((size_t)ss * TB + bb) * N + n;
        } else {
          oi = (size_t)m * N + n;
        }
        C[oi] = v;
      }
}

// ---------------------------------------------------------------------------
// Flash attention. Q,K,V: [B*S, H*DH] head-concat. Grid (S/64, H, B),
// block 128 = 4 waves; wave w owns q rows [blk*64+w*16, +16).
// sK/sV are fragment-ordered so B operands are single 32B LDS reads.
// ---------------------------------------------------------------------------
template <int CAUSAL>
__global__ __launch_bounds__(128) void attn_k(const float* __restrict__ Q,
                                              const float* __restrict__ Kv,
                                              const float* __restrict__ Vv,
                                              float* __restrict__ O) {
  // sK: [e-chunk(2)][n-sub(2)][lane][slot]; sV: [e-sub(4)][lane][slot]
  __shared__ alignas(32) __bf16 sK[4 * 512];
  __shared__ alignas(32) __bf16 sV[4 * 512];
  __shared__ alignas(32) __bf16 sP[4 * 512];   // per-wave A-order P scratch

  const int tid = threadIdx.x;
  const int b = blockIdx.z, h = blockIdx.y;
  const int qblk = blockIdx.x * 64;
  const int w = tid >> 5;
  const int q0 = qblk + w * 16;
  const int lane = tid & 31, half = lane >> 4, l16 = lane & 15;

  // Q fragments: per lane the A layout is one contiguous 16-float run in k
  const float* qbase = Q + ((size_t)b * TS + q0) * TD + h * TDH;
  v16bf aq[2];
#pragma unroll
  for (int c = 0; c < 2; ++c)
#pragma unroll
    for (int g = 0; g < 2; ++g)
#pragma unroll
      for (int j = 0; j < 2; ++j) {
        float4 v = *(const float4*)(qbase + (size_t)l16 * TD + c * 32 +
                                    16 * half + 8 * g + 4 * j);
        float4 vs; vs.x = v.x * 0.125f; vs.y = v.y * 0.125f;
        vs.z = v.z * 0.125f; vs.w = v.w * 0.125f;
        bf4 bb = cvt_bf4(vs);
        const int s0 = 8 * g + 4 * j;
        aq[c][s0 + 0] = bb[0];
        aq[c][s0 + 1] = bb[1];
        aq[c][s0 + 2] = bb[2];
        aq[c][s0 + 3] = bb[3];
      }

  float mrun[8], lrun[8];
  v8f oacc[4] = {};
#pragma unroll
  for (int r = 0; r < 8; ++r) { mrun[r] = -1e30f; lrun[r] = 0.0f; }

  const int nkt = CAUSAL ? (qblk / 32 + 2) : (TS / 32);
  __bf16* myP = sP + w * 512;

  for (int t = 0; t < nkt; ++t) {
    const int k0 = t * 32;
    // cooperative staging of K,V 32x64 tiles, float4 loads, fragment order
#pragma unroll
    for (int i = 0; i < 4; ++i) {
      const int idx4 = i * 128 + tid;          // 512 float4 total
      const int r = idx4 >> 4, e4 = (idx4 & 15) * 4;
      const size_t g = ((size_t)b * TS + k0 + r) * TD + h * TDH + e4;
      float4 kvec = *(const float4*)(Kv + g);
      float4 vvec = *(const float4*)(Vv + g);
      // K^T as B: lane=(r&15)+16*((e&31)>>4), slot=e&15, sub=chunk*2+(r>>4)
      {
        const int ch = e4 >> 5, el = e4 & 31;
        const int ln = (r & 15) + 16 * (el >> 4);
        st_bf4(&sK[(ch * 2 + (r >> 4)) * 512 + ln * 16 + (el & 15)], kvec);
      }
      // V as B: sub=e>>4, lane=(e&15)+16*(r>>4), slot=r&15
      {
        const int sub = e4 >> 4;
        const int ln0 = (e4 & 15) + 16 * (r >> 4);
        __bf16* p = &sV[sub * 512 + ln0 * 16 + (r & 15)];
        bf4 bb = cvt_bf4(vvec);
        p[0]  = bb[0];
        p[16] = bb[1];
        p[32] = bb[2];
        p[48] = bb[3];
      }
    }
    __syncthreads();

    if (!(CAUSAL && k0 > q0 + 15)) {   // wave-uniform -> EXEC all 1s inside
      // scores S = Q * K^T : two 16x16 output frags over the 32 k-columns
      v8f s[2] = {};
#pragma unroll
      for (int nj = 0; nj < 2; ++nj)
#pragma unroll
        for (int ce = 0; ce < 2; ++ce) {
          v16bf bk = ld_frag(&sK[(ce * 2 + nj) * 512]);
          s[nj] = __builtin_amdgcn_wmma_f32_16x16x32_bf16(
              false, aq[ce], false, bk, (short)0, s[nj], false, false);
        }
      if (CAUSAL) {
#pragma unroll
        for (int nj = 0; nj < 2; ++nj)
#pragma unroll
          for (int r = 0; r < 8; ++r) {
            const int ki = k0 + nj * 16 + l16;
            const int qi = q0 + r + 8 * half;
            if (ki > qi) s[nj][r] = -1e30f;
          }
      }
      // online softmax; row reductions across the 16-lane group
      float pmax[8];
#pragma unroll
      for (int r = 0; r < 8; ++r) pmax[r] = fmaxf(s[0][r], s[1][r]);
#pragma unroll
      for (int d = 1; d < 16; d <<= 1)
#pragma unroll
        for (int r = 0; r < 8; ++r)
          pmax[r] = fmaxf(pmax[r], __shfl_xor(pmax[r], d, 32));
      float alpha[8], rsum[8];
#pragma unroll
      for (int r = 0; r < 8; ++r) {
        const float mn = fmaxf(mrun[r], pmax[r]);
        alpha[r] = __expf(mrun[r] - mn);
        mrun[r] = mn;
      }
#pragma unroll
      for (int nj = 0; nj < 2; ++nj)
#pragma unroll
        for (int r = 0; r < 8; ++r) s[nj][r] = __expf(s[nj][r] - mrun[r]);
#pragma unroll
      for (int r = 0; r < 8; ++r) rsum[r] = s[0][r] + s[1][r];
#pragma unroll
      for (int d = 1; d < 16; d <<= 1)
#pragma unroll
        for (int r = 0; r < 8; ++r) rsum[r] += __shfl_xor(rsum[r], d, 32);
#pragma unroll
      for (int r = 0; r < 8; ++r) lrun[r] = lrun[r] * alpha[r] + rsum[r];
#pragma unroll
      for (int c = 0; c < 4; ++c)
#pragma unroll
        for (int r = 0; r < 8; ++r) oacc[c][r] *= alpha[r];

      // P: C-layout -> per-wave LDS in A order (same-wave DS ops in-order)
#pragma unroll
      for (int nj = 0; nj < 2; ++nj)
#pragma unroll
        for (int r = 0; r < 8; ++r) {
          const int ln = (r + 8 * half) + 16 * nj;
          const int sl = l16;
          myP[ln * 16 + sl] = f2bf(s[nj][r]);
        }
      v16bf ap = ld_frag(myP);
      // O += P * V  (K=32, N=64 in 4 chunks)
#pragma unroll
      for (int c = 0; c < 4; ++c) {
        v16bf bv = ld_frag(&sV[c * 512]);
        oacc[c] = __builtin_amdgcn_wmma_f32_16x16x32_bf16(
            false, ap, false, bv, (short)0, oacc[c], false, false);
      }
    }
    __syncthreads();
  }

#pragma unroll
  for (int c = 0; c < 4; ++c)
#pragma unroll
    for (int r = 0; r < 8; ++r) {
      const float inv = 1.0f / lrun[r];
      O[((size_t)b * TS + q0 + r + 8 * half) * TD + h * TDH + c * 16 + l16] =
          oacc[c][r] * inv;
    }
}

// ---------------------------------------------------------------------------
// Fused residual add + LayerNorm over D=1024. One block (256 thr) per row.
// ---------------------------------------------------------------------------
__global__ __launch_bounds__(256) void add_ln_k(const float* __restrict__ X,
                                                const float* __restrict__ R,
                                                const float* __restrict__ g,
                                                const float* __restrict__ be,
                                                float* __restrict__ Y) {
  __shared__ float red[256];
  const int row = blockIdx.x, tid = threadIdx.x;
  const int d = tid * 4;
  float4 xv = *(const float4*)(X + (size_t)row * TD + d);
  float4 rv = *(const float4*)(R + (size_t)row * TD + d);
  float v[4] = { xv.x + rv.x, xv.y + rv.y, xv.z + rv.z, xv.w + rv.w };
  float s = v[0] + v[1] + v[2] + v[3];
  red[tid] = s; __syncthreads();
  for (int dd = 128; dd > 0; dd >>= 1) { if (tid < dd) red[tid] += red[tid + dd]; __syncthreads(); }
  const float mean = red[0] / (float)TD;
  __syncthreads();
  float s2 = 0.0f;
#pragma unroll
  for (int i = 0; i < 4; ++i) { const float dv = v[i] - mean; s2 += dv * dv; }
  red[tid] = s2; __syncthreads();
  for (int dd = 128; dd > 0; dd >>= 1) { if (tid < dd) red[tid] += red[tid + dd]; __syncthreads(); }
  const float inv = rsqrtf(red[0] / (float)TD + 1e-5f);
  float4 gv = *(const float4*)(g + d);
  float4 bv = *(const float4*)(be + d);
  float4 o;
  o.x = gv.x * (v[0] - mean) * inv + bv.x;
  o.y = gv.y * (v[1] - mean) * inv + bv.y;
  o.z = gv.z * (v[2] - mean) * inv + bv.z;
  o.w = gv.w * (v[3] - mean) * inv + bv.w;
  *(float4*)(Y + (size_t)row * TD + d) = o;
}

// ---------------------------------------------------------------------------
// Host orchestration
// ---------------------------------------------------------------------------
static inline void gemm(const float* A, const float* B, const float* bias,
                        float* C, int M, int N, int K, int flags,
                        hipStream_t st) {
  dim3 grid(M / 128, N / 128);
  switch (flags) {
    case F_RELU: gemm_k<F_RELU><<<grid, 256, 0, st>>>(A, B, bias, C, M, N, K); break;
    case F_QKV:  gemm_k<F_QKV ><<<grid, 256, 0, st>>>(A, B, bias, C, M, N, K); break;
    case F_SWZ:  gemm_k<F_SWZ ><<<grid, 256, 0, st>>>(A, B, bias, C, M, N, K); break;
    default:     gemm_k<0     ><<<grid, 256, 0, st>>>(A, B, bias, C, M, N, K); break;
  }
}

extern "C" void kernel_launch(void* const* d_in, const int* in_sizes, int n_in,
                              void* d_out, int out_size, void* d_ws, size_t ws_size,
                              hipStream_t stream) {
  const int*   input  = (const int*)d_in[0];
  const float* PE     = (const float*)d_in[2];
  const float* embedW = (const float*)d_in[3];
  const float* eWq = (const float*)d_in[4],  *ebq = (const float*)d_in[5];
  const float* eWk = (const float*)d_in[6],  *ebk = (const float*)d_in[7];
  const float* eWv = (const float*)d_in[8],  *ebv = (const float*)d_in[9];
  const float* eWo = (const float*)d_in[10], *ebo = (const float*)d_in[11];
  const float* dWq0 = (const float*)d_in[12], *dbq0 = (const float*)d_in[13];
  const float* dWk0 = (const float*)d_in[14], *dbk0 = (const float*)d_in[15];
  const float* dWv0 = (const float*)d_in[16], *dbv0 = (const float*)d_in[17];
  const float* dWo0 = (const float*)d_in[18], *dbo0 = (const float*)d_in[19];
  const float* dWq1 = (const float*)d_in[20], *dbq1 = (const float*)d_in[21];
  const float* dWk1 = (const float*)d_in[22], *dbk1 = (const float*)d_in[23];
  const float* dWv1 = (const float*)d_in[24], *dbv1 = (const float*)d_in[25];
  const float* dWo1 = (const float*)d_in[26], *dbo1 = (const float*)d_in[27];
  const float* eg1 = (const float*)d_in[28], *eg2 = (const float*)d_in[29];
  const float* dg0 = (const float*)d_in[30], *dg1 = (const float*)d_in[31];
  const float* dg2 = (const float*)d_in[32];
  const float* ebe1 = (const float*)d_in[33], *ebe2 = (const float*)d_in[34];
  const float* dbe0 = (const float*)d_in[35], *dbe1 = (const float*)d_in[36];
  const float* dbe2 = (const float*)d_in[37];
  const float* eW1 = (const float*)d_in[38], *eb1 = (const float*)d_in[39];
  const float* eW2 = (const float*)d_in[40], *eb2 = (const float*)d_in[41];
  const float* dW1 = (const float*)d_in[42], *db1 = (const float*)d_in[43];
  const float* dW2 = (const float*)d_in[44], *db2 = (const float*)d_in[45];
  const float* Wl = (const float*)d_in[46], *bl = (const float*)d_in[47];

  const size_t MD = (size_t)TM * TD;
  float* ws  = (float*)d_ws;
  float* ENC = ws;
  float* DEC = ws + 1 * MD;
  float* Qb  = ws + 2 * MD;
  float* Kb  = ws + 3 * MD;
  float* Vb  = ws + 4 * MD;
  float* ATT = ws + 5 * MD;
  float* T2  = ws + 6 * MD;
  float* T1  = ws + 7 * MD;                 // [TM, DFF]

  const size_t WA = (size_t)TH * TD * TDH;
  const size_t WF = (size_t)TD * TDFF;

  embed_k<<<TM, 256, 0, stream>>>(input, PE, embedW, ENC, DEC);

  // ---- encoder ----
  for (int l = 0; l < TL; ++l) {
    gemm(ENC, eWq + l * WA, ebq + l * TD, Qb, TM, TD, TD, F_QKV, stream);
    gemm(ENC, eWk + l * WA, ebk + l * TD, Kb, TM, TD, TD, F_QKV, stream);
    gemm(ENC, eWv + l * WA, ebv + l * TD, Vb, TM, TD, TD, F_QKV, stream);
    attn_k<0><<<dim3(TS / 64, TH, TB), 128, 0, stream>>>(Qb, Kb, Vb, ATT);
    gemm(ATT, eWo + l * (size_t)TD * TD, ebo + l * TD, T2, TM, TD, TD, 0, stream);
    add_ln_k<<<TM, 256, 0, stream>>>(ENC, T2, eg1 + l * TD, ebe1 + l * TD, ENC);
    gemm(ENC, eW1 + l * WF, eb1 + l * TDFF, T1, TM, TDFF, TD, F_RELU, stream);
    gemm(T1, eW2 + l * WF, eb2 + l * TD, T2, TM, TD, TDFF, 0, stream);
    add_ln_k<<<TM, 256, 0, stream>>>(ENC, T2, eg2 + l * TD, ebe2 + l * TD, ENC);
  }

  // ---- decoder ----
  for (int l = 0; l < TL; ++l) {
    gemm(DEC, dWq0 + l * WA, dbq0 + l * TD, Qb, TM, TD, TD, F_QKV, stream);
    gemm(DEC, dWk0 + l * WA, dbk0 + l * TD, Kb, TM, TD, TD, F_QKV, stream);
    gemm(DEC, dWv0 + l * WA, dbv0 + l * TD, Vb, TM, TD, TD, F_QKV, stream);
    attn_k<1><<<dim3(TS / 64, TH, TB), 128, 0, stream>>>(Qb, Kb, Vb, ATT);
    gemm(ATT, dWo0 + l * (size_t)TD * TD, dbo0 + l * TD, T2, TM, TD, TD, 0, stream);
    add_ln_k<<<TM, 256, 0, stream>>>(DEC, T2, dg0 + l * TD, dbe0 + l * TD, DEC);
    gemm(DEC, dWq1 + l * WA, dbq1 + l * TD, Qb, TM, TD, TD, F_QKV, stream);
    gemm(ENC, dWk1 + l * WA, dbk1 + l * TD, Kb, TM, TD, TD, F_QKV, stream);
    gemm(ENC, dWv1 + l * WA, dbv1 + l * TD, Vb, TM, TD, TD, F_QKV, stream);
    attn_k<0><<<dim3(TS / 64, TH, TB), 128, 0, stream>>>(Qb, Kb, Vb, ATT);
    gemm(ATT, dWo1 + l * (size_t)TD * TD, dbo1 + l * TD, T2, TM, TD, TD, 0, stream);
    add_ln_k<<<TM, 256, 0, stream>>>(DEC, T2, dg1 + l * TD, dbe1 + l * TD, DEC);
    gemm(DEC, dW1 + l * WF, db1 + l * TDFF, T1, TM, TDFF, TD, F_RELU, stream);
    gemm(T1, dW2 + l * WF, db2 + l * TD, T2, TM, TD, TDFF, 0, stream);
    add_ln_k<<<TM, 256, 0, stream>>>(DEC, T2, dg2 + l * TD, dbe2 + l * TD, DEC);
  }

  // ---- final vocab projection -> d_out as [S,B,V] ----
  gemm(DEC, Wl, bl, (float*)d_out, TM, TV, TD, F_SWZ, stream);
}
